// TripletLoss_67070209294941
// MI455X (gfx1250) — matrix operand order repaired and checked by reading, not compiled, not run on hardware
//
#include <hip/hip_runtime.h>
#include <hip/hip_bf16.h>
#include <stdint.h>

typedef __attribute__((ext_vector_type(16))) __bf16 v16bf;
typedef __attribute__((ext_vector_type(8)))  __bf16 v8bf;
typedef __attribute__((ext_vector_type(8)))  float  v8f;

#define FEAT_K 256
#define NROWS  16384
#define HALF   8192
#define NBLK   (HALF / 16)
#define MARGIN 100.0f

// ---- gfx1250 async global->LDS copy (ASYNCcnt-tracked, no VGPR round trip) ----
__device__ __forceinline__ void async_copy_b128(uint32_t lds_addr, const void* gptr) {
    asm volatile("global_load_async_to_lds_b128 %0, %1, off"
                 :: "v"(lds_addr), "v"(gptr)
                 : "memory");
}
__device__ __forceinline__ void wait_async_all() {
    asm volatile("s_wait_asynccnt 0x0" ::: "memory");
}

// ---------------------------------------------------------------------------
// Kernel 1: split f32 feats -> bf16 hi/lo pair, and compute row norms |x|^2.
// ---------------------------------------------------------------------------
__global__ __launch_bounds__(256) void tl_prep_kernel(const float* __restrict__ feats,
                                                      __bf16* __restrict__ hi,
                                                      __bf16* __restrict__ lo,
                                                      float* __restrict__ nrm)
{
    const int row = blockIdx.x;
    const int k   = threadIdx.x;
    const size_t idx = (size_t)row * FEAT_K + k;
    float x = feats[idx];
    __bf16 hv = (__bf16)x;                 // RNE to bf16 (top ~8 mantissa bits)
    __bf16 lv = (__bf16)(x - (float)hv);   // residual (next ~8 bits)
    hi[idx] = hv;
    lo[idx] = lv;

    __shared__ float sb[256];
    sb[k] = x * x;
    __syncthreads();
    for (int off = 128; off > 0; off >>= 1) {
        if (k < off) sb[k] += sb[k + off];
        __syncthreads();
    }
    if (k == 0) nrm[row] = sb[0];
}

// ---------------------------------------------------------------------------
// Kernel 2: fused GEMM + per-column reduction (bf16x3 split ~= fp32 precision).
//   acc_ij = n_i . x_j - |n_i|^2/2 ; R[j] = min_i (s*acc) with s=+1 (normal
//   columns -> max distance) or s=-1 (abnormal -> min distance), stored as
//   s*red so that extreme d^2 = norm[j] - 2*R[j] in both halves.
//   One wave owns 16 columns (A register-resident, reused 512x). B blocks
//   (16 normal rows, hi+lo) double-buffered in LDS via async global->LDS DMA.
// ---------------------------------------------------------------------------
__global__ __launch_bounds__(256) void tl_wmma_kernel(const __bf16* __restrict__ hi,
                                                      const __bf16* __restrict__ lo,
                                                      const float* __restrict__ nrm,
                                                      float* __restrict__ R)
{
    // [buffer][hi/lo][row][k]; +8 bf16 row pad keeps 16B alignment & spreads banks
    __shared__ __bf16 Bbuf[2][2][16][FEAT_K + 8];

    const int lane = threadIdx.x & 31;
    const int wave = threadIdx.x >> 5;
    const int h    = lane >> 4;             // half-wave (K-split of operands)
    const int t    = lane & 15;
    const int jbase = blockIdx.x * 128 + wave * 16;
    const float s  = (jbase < HALF) ? 1.0f : -1.0f;

    // ---- A fragments (16 j-rows, 8 K-chunks, hi & lo) -> registers.
    // ISA 16-bit A layout per lane: K = 32c + 8h + e (e<8), K = 32c + 16 + 8h + (e-8).
    v16bf aH[8], aL[8];
    {
        const size_t rowoff = (size_t)(jbase + t) * FEAT_K;
        #pragma unroll
        for (int c = 0; c < 8; ++c) {
            const int k0 = c * 32 + h * 8;
            v8bf h0 = *(const v8bf*)(hi + rowoff + k0);
            v8bf h1 = *(const v8bf*)(hi + rowoff + k0 + 16);
            v8bf l0 = *(const v8bf*)(lo + rowoff + k0);
            v8bf l1 = *(const v8bf*)(lo + rowoff + k0 + 16);
            aH[c] = __builtin_shufflevector(h0, h1, 0,1,2,3,4,5,6,7,8,9,10,11,12,13,14,15);
            aL[c] = __builtin_shufflevector(l0, l1, 0,1,2,3,4,5,6,7,8,9,10,11,12,13,14,15);
        }
    }

    float red[8];
    #pragma unroll
    for (int v = 0; v < 8; ++v) red[v] = 3.402823466e38f;

    // Cooperative stage: 256 threads x 64B = 16KB block (hi+lo), 4 async b128 each.
    const int cr = threadIdx.x >> 4;          // row   0..15
    const int ck = (threadIdx.x & 15) * 16;   // col   (bf16 elems)

#define STAGE_BLOCK(mrow, q)                                                    \
    do {                                                                        \
        const size_t g = (size_t)((mrow) + cr) * FEAT_K + ck;                   \
        uint32_t dh = (uint32_t)(uintptr_t)&Bbuf[(q)][0][cr][ck];               \
        uint32_t dl = (uint32_t)(uintptr_t)&Bbuf[(q)][1][cr][ck];               \
        async_copy_b128(dh,      hi + g);                                       \
        async_copy_b128(dh + 16, hi + g + 8);                                   \
        async_copy_b128(dl,      lo + g);                                       \
        async_copy_b128(dl + 16, lo + g + 8);                                   \
    } while (0)

    // Prologue: stage block 0, prefetch first norm.
    STAGE_BLOCK(0, 0);
    float cn = -0.5f * nrm[t];
    wait_async_all();
    __syncthreads();

    for (int mb = 0; mb < NBLK; ++mb) {
        const int p = mb & 1;

        // Stage next block into the other buffer while this one computes.
        if (mb + 1 < NBLK) STAGE_BLOCK((mb + 1) * 16, p ^ 1);

        // Pipelined accumulator seed: C[v] = -|n_i|^2/2, i = mb*16 + (lane%16).
        const float cinit = cn;
        if (mb + 1 < NBLK) cn = -0.5f * nrm[(mb + 1) * 16 + t];

        v8f acc;
        #pragma unroll
        for (int v = 0; v < 8; ++v) acc[v] = cinit;

        // 8 K-chunks x 3 split WMMAs: acc += Ah*Bh + Ah*Bl + Al*Bh
        #pragma unroll
        for (int c = 0; c < 8; ++c) {
            const int k0 = c * 32 + h * 8;    // ISA 16-bit B layout: K = 32c + 8h + e (e<8), 32c + 16 + 8h + (e-8)
            v8bf b0 = *(const v8bf*)&Bbuf[p][0][t][k0];
            v8bf b1 = *(const v8bf*)&Bbuf[p][0][t][k0 + 16];
            v8bf c0 = *(const v8bf*)&Bbuf[p][1][t][k0];
            v8bf c1 = *(const v8bf*)&Bbuf[p][1][t][k0 + 16];
            v16bf bH = __builtin_shufflevector(b0, b1, 0,1,2,3,4,5,6,7,8,9,10,11,12,13,14,15);
            v16bf bL = __builtin_shufflevector(c0, c1, 0,1,2,3,4,5,6,7,8,9,10,11,12,13,14,15);
            acc = __builtin_amdgcn_wmma_f32_16x16x32_bf16(false, aH[c], false, bH, (short)0, acc, false, false);
            acc = __builtin_amdgcn_wmma_f32_16x16x32_bf16(false, aH[c], false, bL, (short)0, acc, false, false);
            acc = __builtin_amdgcn_wmma_f32_16x16x32_bf16(false, aL[c], false, bH, (short)0, acc, false, false);
        }

        // Branchless running extreme over this block's 16 i values.
        #pragma unroll
        for (int v = 0; v < 8; ++v) red[v] = fminf(red[v], s * acc[v]);

        // Our async writes into buffer p^1 must be in LDS before anyone reads it.
        wait_async_all();
        __syncthreads();
    }
#undef STAGE_BLOCK

    // Cross-lane min over the 16 lanes of each half-wave (same j set per half).
    #pragma unroll
    for (int v = 0; v < 8; ++v) {
        float r = red[v];
        r = fminf(r, __shfl_xor(r, 1, 32));
        r = fminf(r, __shfl_xor(r, 2, 32));
        r = fminf(r, __shfl_xor(r, 4, 32));
        r = fminf(r, __shfl_xor(r, 8, 32));
        red[v] = r;
    }
    // D layout: (VGPR v, half h) -> column j = jbase + v + 8h.
    if (t == 0) {
        #pragma unroll
        for (int v = 0; v < 8; ++v)
            R[jbase + 8 * h + v] = s * red[v];
    }
}

// ---------------------------------------------------------------------------
// Kernel 3: scalar epilogue. Extreme squared distance per column:
//   d2_j = max(0, |x_j|^2 - 2*R[j]);  loss = mean(d_norm) + mean(relu(M-d_abn)).
// ---------------------------------------------------------------------------
__global__ __launch_bounds__(256) void tl_final_kernel(const float* __restrict__ R,
                                                       const float* __restrict__ nrm,
                                                       float* __restrict__ out)
{
    float acc = 0.0f;
    for (int j = threadIdx.x; j < NROWS; j += 256) {
        float d = sqrtf(fmaxf(nrm[j] - 2.0f * R[j], 0.0f));
        acc += (j < HALF) ? d : fmaxf(MARGIN - d, 0.0f);
    }
    __shared__ float sb[256];
    sb[threadIdx.x] = acc;
    __syncthreads();
    for (int off = 128; off > 0; off >>= 1) {
        if ((int)threadIdx.x < off) sb[threadIdx.x] += sb[threadIdx.x + off];
        __syncthreads();
    }
    if (threadIdx.x == 0) out[0] = sb[0] * (1.0f / (float)HALF);
}

// ---------------------------------------------------------------------------
extern "C" void kernel_launch(void* const* d_in, const int* in_sizes, int n_in,
                              void* d_out, int out_size, void* d_ws, size_t ws_size,
                              hipStream_t stream)
{
    (void)in_sizes; (void)n_in; (void)out_size; (void)ws_size;
    const float* feats = (const float*)d_in[0];

    // Workspace layout (~16.1 MB):
    char* ws = (char*)d_ws;
    __bf16* hi  = (__bf16*)ws;                                        // 8 MB
    __bf16* lo  = (__bf16*)(ws + (size_t)NROWS * FEAT_K * 2);         // 8 MB
    float*  nrm = (float*)(ws + (size_t)NROWS * FEAT_K * 4);          // 64 KB
    float*  R   = nrm + NROWS;                                        // 64 KB

    tl_prep_kernel<<<NROWS, 256, 0, stream>>>(feats, hi, lo, nrm);
    tl_wmma_kernel<<<NROWS / 128, 256, 0, stream>>>(hi, lo, nrm, R);
    tl_final_kernel<<<1, 256, 0, stream>>>(R, nrm, (float*)d_out);
}
